// histoGIN_31937376813167
// MI455X (gfx1250) — compile-verified, matrix operand order checked by reading, not run on hardware
//
#include <hip/hip_runtime.h>
#include <hip/hip_bf16.h>

typedef __attribute__((ext_vector_type(16))) _Float16 v16h;
typedef __attribute__((ext_vector_type(8)))  _Float16 v8h;
typedef __attribute__((ext_vector_type(4)))  _Float16 v4h;
typedef __attribute__((ext_vector_type(8)))  float    v8f;

static constexpr int EMB      = 128;
static constexpr int HID      = 64;
static constexpr int N_GRAPHS = 64;

// ---------------------------------------------------------------------------
// Utility kernels
// ---------------------------------------------------------------------------
__global__ void k_zero(float* __restrict__ p, int n) {
    int i = blockIdx.x * blockDim.x + threadIdx.x;
    if (i < n) p[i] = 0.0f;
}

__global__ void k_copy4(const float4* __restrict__ in, float4* __restrict__ out, int n4) {
    int i = blockIdx.x * blockDim.x + threadIdx.x;
    if (i < n4) out[i] = in[i];
}

__global__ void k_cast4(const float* __restrict__ in, _Float16* __restrict__ out, int n4) {
    int i = blockIdx.x * blockDim.x + threadIdx.x;
    if (i < n4) {
        float4 v = ((const float4*)in)[i];
        v4h o;
        o[0] = (_Float16)v.x; o[1] = (_Float16)v.y;
        o[2] = (_Float16)v.z; o[3] = (_Float16)v.w;
        ((v4h*)out)[i] = o;
    }
}

// ---------------------------------------------------------------------------
// Pack weights [K,64] f32 into WMMA B-fragment order (f16):
//   P[(((ks*4 + nt)*32 + lane)*16) + e]  =  W[k, nt*16 + r]
//   k = ks*32 + (e<8 ? h*8+e : 16 + h*8 + (e-8)),  h=lane>>4, r=lane&15
// ---------------------------------------------------------------------------
template<int K>
__global__ void k_pack_w(const float* __restrict__ W, _Float16* __restrict__ P) {
    int i = blockIdx.x * blockDim.x + threadIdx.x;
    if (i >= K * 64) return;
    int e    = i & 15;
    int lane = (i >> 4) & 31;
    int nt   = (i >> 9) & 3;
    int ks   = i >> 11;
    int h = lane >> 4, r = lane & 15;
    int k = ks * 32 + (e < 8 ? h * 8 + e : 16 + h * 8 + (e - 8));
    P[i] = (_Float16)W[k * 64 + nt * 16 + r];
}

// ---------------------------------------------------------------------------
// Edge scatter-add:  agg[dst] += feat[src]   (agg pre-initialized with feat)
// ---------------------------------------------------------------------------
template<int F>
__global__ void k_edge_scatter(const float* __restrict__ feat,
                               const long long* __restrict__ src,
                               const long long* __restrict__ dst,
                               float* __restrict__ agg, int nE) {
    constexpr int GP = F / 4;
    long long t = (long long)blockIdx.x * blockDim.x + threadIdx.x;
    long long e = t / GP;
    int       g = (int)(t % GP);
    if (e >= nE) return;
    int s = (int)src[e];
    int d = (int)dst[e];
    const float4 v = *(const float4*)(feat + (size_t)s * F + g * 4);
    float* out = agg + (size_t)d * F + g * 4;
    atomicAdd(out + 0, v.x);
    atomicAdd(out + 1, v.y);
    atomicAdd(out + 2, v.z);
    atomicAdd(out + 3, v.w);
}

// ---------------------------------------------------------------------------
// Persistent WMMA GEMM:  C[N,64] = act( A[N,K] @ W[K,64] + bias )
//  - K=64 : 8 B fragments hoisted into VGPRs (fits without spilling)
//  - K=128: B staged in LDS once per block; fragments re-read per tile
//           (opaque offset defeats LICM so nothing spills to scratch)
// ---------------------------------------------------------------------------
template<int K, bool RELU, bool STORE_HALF>
__global__ void k_gemm_wmma(const _Float16* __restrict__ A,    // [N, K]
                            const _Float16* __restrict__ P,    // packed B fragments
                            const float*    __restrict__ bias, // [64]
                            float*          __restrict__ Cf,   // [N, 64]
                            _Float16*       __restrict__ Ch,   // [N, 64]
                            int ntiles) {
    constexpr int KS = K / 32;
    const int lane   = threadIdx.x & 31;
    const int wave   = threadIdx.x >> 5;
    const int nwaves = (gridDim.x * blockDim.x) >> 5;
    const int wid    = blockIdx.x * (blockDim.x >> 5) + wave;
    const int h = lane >> 4;
    const int r = lane & 15;

    float bv[4];
#pragma unroll
    for (int nt = 0; nt < 4; ++nt) bv[nt] = bias[nt * 16 + r];

    if constexpr (KS <= 2) {
        // ---- register-hoisted B fragments (64 VGPRs) ----
        v16h bfrag[KS][4];
#pragma unroll
        for (int ks = 0; ks < KS; ++ks)
#pragma unroll
            for (int nt = 0; nt < 4; ++nt) {
                const v8h* bp = (const v8h*)(P + (size_t)((ks * 4 + nt) * 32 + lane) * 16);
                ((v8h*)&bfrag[ks][nt])[0] = bp[0];
                ((v8h*)&bfrag[ks][nt])[1] = bp[1];
            }
        for (int tile = wid; tile < ntiles; tile += nwaves) {
            v8f acc[4] = {};
            const _Float16* arow = A + (size_t)(tile * 16 + r) * K + h * 8;
#pragma unroll
            for (int ks = 0; ks < KS; ++ks) {
                v16h a;
                ((v8h*)&a)[0] = *(const v8h*)(arow + ks * 32);
                ((v8h*)&a)[1] = *(const v8h*)(arow + ks * 32 + 16);
#pragma unroll
                for (int nt = 0; nt < 4; ++nt)
                    acc[nt] = __builtin_amdgcn_wmma_f32_16x16x32_f16(
                        false, a, false, bfrag[ks][nt], (short)0, acc[nt], false, false);
            }
#pragma unroll
            for (int nt = 0; nt < 4; ++nt)
#pragma unroll
                for (int v = 0; v < 8; ++v) {
                    const int row = tile * 16 + v + h * 8;
                    float val = acc[nt][v] + bv[nt];
                    if (RELU) val = fmaxf(val, 0.0f);
                    const size_t idx = (size_t)row * 64 + nt * 16 + r;
                    if (STORE_HALF) Ch[idx] = (_Float16)val;
                    else            Cf[idx] = val;
                }
        }
    } else {
        // ---- LDS-staged B (16 KB), re-read per tile ----
        __shared__ _Float16 Bs[K * 64];
        for (int i = threadIdx.x; i < K * 64 / 8; i += blockDim.x)
            ((v8h*)Bs)[i] = ((const v8h*)P)[i];
        __syncthreads();

        for (int tile = wid; tile < ntiles; tile += nwaves) {
            int ob = 0;
            asm volatile("" : "+v"(ob));   // opaque 0: keep ds reads in-loop (no LICM/spill)
            v8f acc[4] = {};
            const _Float16* arow = A + (size_t)(tile * 16 + r) * K + h * 8;
#pragma unroll
            for (int ks = 0; ks < KS; ++ks) {
                v16h a;
                ((v8h*)&a)[0] = *(const v8h*)(arow + ks * 32);
                ((v8h*)&a)[1] = *(const v8h*)(arow + ks * 32 + 16);
#pragma unroll
                for (int nt = 0; nt < 4; ++nt) {
                    const v8h* bp = (const v8h*)(Bs + ob + (size_t)((ks * 4 + nt) * 32 + lane) * 16);
                    v16h b;
                    ((v8h*)&b)[0] = bp[0];
                    ((v8h*)&b)[1] = bp[1];
                    acc[nt] = __builtin_amdgcn_wmma_f32_16x16x32_f16(
                        false, a, false, b, (short)0, acc[nt], false, false);
                }
            }
#pragma unroll
            for (int nt = 0; nt < 4; ++nt)
#pragma unroll
                for (int v = 0; v < 8; ++v) {
                    const int row = tile * 16 + v + h * 8;
                    float val = acc[nt][v] + bv[nt];
                    if (RELU) val = fmaxf(val, 0.0f);
                    const size_t idx = (size_t)row * 64 + nt * 16 + r;
                    if (STORE_HALF) Ch[idx] = (_Float16)val;
                    else            Cf[idx] = val;
                }
        }
    }
}

// ---------------------------------------------------------------------------
// Mean-pool stage 1: atomic segment sums + counts
// ---------------------------------------------------------------------------
__global__ void k_pool(const float* __restrict__ h,
                       const long long* __restrict__ batch,
                       float* __restrict__ sums,    // [G, HID]
                       float* __restrict__ counts,  // [G]
                       int nNodes) {
    int t = blockIdx.x * blockDim.x + threadIdx.x;
    int node = t >> 6;
    int f    = t & 63;
    if (node >= nNodes) return;
    int g = (int)batch[node];
    atomicAdd(&sums[g * HID + f], h[(size_t)node * HID + f]);
    if (f == 0) atomicAdd(&counts[g], 1.0f);
}

// ---------------------------------------------------------------------------
// Head: pooled = sums/count ; fc1 (64->32) ; pred (32->1) ; sigmoid
// ---------------------------------------------------------------------------
__global__ void k_head(const float* __restrict__ sums,
                       const float* __restrict__ counts,
                       const float* __restrict__ fc1W,  // [64,32]
                       const float* __restrict__ fc1b,  // [32]
                       const float* __restrict__ predW, // [32]
                       const float* __restrict__ predb, // [1]
                       float* __restrict__ out) {
    int g = threadIdx.x;
    if (g >= N_GRAPHS) return;
    float inv = 1.0f / fmaxf(counts[g], 1.0f);
    float pooled[HID];
#pragma unroll
    for (int f = 0; f < HID; ++f) pooled[f] = sums[g * HID + f] * inv;
    float s = predb[0];
    for (int j = 0; j < 32; ++j) {
        float t = fc1b[j];
#pragma unroll
        for (int f = 0; f < HID; ++f) t += pooled[f] * fc1W[f * 32 + j];
        s += t * predW[j];
    }
    out[g] = 1.0f / (1.0f + __expf(-s));
}

// ---------------------------------------------------------------------------
// Launch
// ---------------------------------------------------------------------------
extern "C" void kernel_launch(void* const* d_in, const int* in_sizes, int n_in,
                              void* d_out, int out_size, void* d_ws, size_t ws_size,
                              hipStream_t stream) {
    const float*     x     = (const float*)d_in[0];
    const long long* ei    = (const long long*)d_in[1];
    const long long* batch = (const long long*)d_in[2];
    const float *c1W1 = (const float*)d_in[3],  *c1b1 = (const float*)d_in[4];
    const float *c1W2 = (const float*)d_in[5],  *c1b2 = (const float*)d_in[6];
    const float *c2W1 = (const float*)d_in[7],  *c2b1 = (const float*)d_in[8];
    const float *c2W2 = (const float*)d_in[9],  *c2b2 = (const float*)d_in[10];
    const float *c3W1 = (const float*)d_in[11], *c3b1 = (const float*)d_in[12];
    const float *c3W2 = (const float*)d_in[13], *c3b2 = (const float*)d_in[14];
    const float *fc1W = (const float*)d_in[15], *fc1b = (const float*)d_in[16];
    const float *prW  = (const float*)d_in[17], *prb  = (const float*)d_in[18];

    const int N  = in_sizes[0] / EMB;        // 100000
    const int nE = in_sizes[1] / 2;          // 1600000
    const long long* src = ei;
    const long long* dst = ei + nE;

    // ---- workspace carve-out (256B aligned) ----
    char*  ws  = (char*)d_ws;
    size_t off = 0;
    auto carve = [&](size_t bytes) -> char* {
        char* p = ws + off;
        off += (bytes + 255) & ~(size_t)255;
        return p;
    };
    float*    agg   = (float*)   carve((size_t)N * EMB * sizeof(float));
    _Float16* aggh  = (_Float16*)carve((size_t)N * EMB * sizeof(_Float16));
    _Float16* t1    = (_Float16*)carve((size_t)N * HID * sizeof(_Float16));
    float*    hbuf  = (float*)   carve((size_t)N * HID * sizeof(float));
    _Float16* w1p   = (_Float16*)carve((size_t)EMB * HID * sizeof(_Float16));
    _Float16* w2p   = (_Float16*)carve((size_t)HID * HID * sizeof(_Float16));
    _Float16* w3p   = (_Float16*)carve((size_t)HID * HID * sizeof(_Float16));
    _Float16* w4p   = (_Float16*)carve((size_t)HID * HID * sizeof(_Float16));
    _Float16* w5p   = (_Float16*)carve((size_t)HID * HID * sizeof(_Float16));
    _Float16* w6p   = (_Float16*)carve((size_t)HID * HID * sizeof(_Float16));
    float*    sums  = (float*)   carve((size_t)N_GRAPHS * HID * sizeof(float));
    float*    cnts  = (float*)   carve((size_t)N_GRAPHS * sizeof(float));

    const int T = 256;
    auto blocks = [&](long long n) { return (int)((n + T - 1) / T); };

    // pack weights into B-fragment order (cast + permute, one shot)
    k_pack_w<EMB><<<blocks(EMB * HID), T, 0, stream>>>(c1W1, w1p);
    k_pack_w<HID><<<blocks(HID * HID), T, 0, stream>>>(c1W2, w2p);
    k_pack_w<HID><<<blocks(HID * HID), T, 0, stream>>>(c2W1, w3p);
    k_pack_w<HID><<<blocks(HID * HID), T, 0, stream>>>(c2W2, w4p);
    k_pack_w<HID><<<blocks(HID * HID), T, 0, stream>>>(c3W1, w5p);
    k_pack_w<HID><<<blocks(HID * HID), T, 0, stream>>>(c3W2, w6p);

    // zero pool accumulators (ws is not re-zeroed by the harness)
    k_zero<<<blocks(N_GRAPHS * HID + N_GRAPHS), T, 0, stream>>>(sums, N_GRAPHS * HID + N_GRAPHS);

    const int ntiles = (N + 15) / 16;   // 6250
    const int GB = 512;                 // persistent GEMM: 512 blocks x 4 waves
    const int GT = 128;

    // -------- layer 1 (K = 128) --------
    k_copy4<<<blocks((long long)N * EMB / 4), T, 0, stream>>>((const float4*)x, (float4*)agg, N * EMB / 4);
    k_edge_scatter<EMB><<<blocks((long long)nE * (EMB / 4)), T, 0, stream>>>(x, src, dst, agg, nE);
    k_cast4<<<blocks((long long)N * EMB / 4), T, 0, stream>>>(agg, aggh, N * EMB / 4);
    k_gemm_wmma<EMB, true, true ><<<GB, GT, 0, stream>>>(aggh, w1p, c1b1, nullptr, t1, ntiles);
    k_gemm_wmma<HID, true, false><<<GB, GT, 0, stream>>>(t1,   w2p, c1b2, hbuf, nullptr, ntiles);

    // -------- layer 2 (K = 64) --------
    k_copy4<<<blocks((long long)N * HID / 4), T, 0, stream>>>((const float4*)hbuf, (float4*)agg, N * HID / 4);
    k_edge_scatter<HID><<<blocks((long long)nE * (HID / 4)), T, 0, stream>>>(hbuf, src, dst, agg, nE);
    k_cast4<<<blocks((long long)N * HID / 4), T, 0, stream>>>(agg, aggh, N * HID / 4);
    k_gemm_wmma<HID, true, true ><<<GB, GT, 0, stream>>>(aggh, w3p, c2b1, nullptr, t1, ntiles);
    k_gemm_wmma<HID, true, false><<<GB, GT, 0, stream>>>(t1,   w4p, c2b2, hbuf, nullptr, ntiles);

    // -------- layer 3 (K = 64, no outer ReLU) --------
    k_copy4<<<blocks((long long)N * HID / 4), T, 0, stream>>>((const float4*)hbuf, (float4*)agg, N * HID / 4);
    k_edge_scatter<HID><<<blocks((long long)nE * (HID / 4)), T, 0, stream>>>(hbuf, src, dst, agg, nE);
    k_cast4<<<blocks((long long)N * HID / 4), T, 0, stream>>>(agg, aggh, N * HID / 4);
    k_gemm_wmma<HID, true,  true ><<<GB, GT, 0, stream>>>(aggh, w5p, c3b1, nullptr, t1, ntiles);
    k_gemm_wmma<HID, false, false><<<GB, GT, 0, stream>>>(t1,   w6p, c3b2, hbuf, nullptr, ntiles);

    // -------- pool + head --------
    k_pool<<<blocks((long long)N * HID), T, 0, stream>>>(hbuf, batch, sums, cnts, N);
    k_head<<<1, 64, 0, stream>>>(sums, cnts, fc1W, fc1b, prW, prb, (float*)d_out);
}
